// FPModule_34849364640186
// MI455X (gfx1250) — compile-verified, matrix operand order checked by reading, not run on hardware
//
#include <hip/hip_runtime.h>
#include <stdint.h>

typedef __attribute__((ext_vector_type(16))) __bf16 v16bf;
typedef __attribute__((ext_vector_type(8)))  float  v8f;

struct Pack8 { unsigned int u[8]; };

static __device__ inline unsigned short f32_to_bf16(float f) {
    unsigned int x = __builtin_bit_cast(unsigned int, f);
    x += 0x7FFFu + ((x >> 16) & 1u);   // round-to-nearest-even
    return (unsigned short)(x >> 16);
}

static __device__ inline unsigned int pack_bf16x2(float lo, float hi) {
    return (unsigned int)f32_to_bf16(lo) | ((unsigned int)f32_to_bf16(hi) << 16);
}

// CDNA5 async copy: global -> LDS, 16 bytes per lane, tracked by ASYNCcnt.
static __device__ inline void async_copy_b128(const unsigned short* lds_dst,
                                              const unsigned short* gsrc) {
    unsigned lds_off = (unsigned)(uintptr_t)lds_dst;   // LDS aperture: low 32 bits
    asm volatile("global_load_async_to_lds_b128 %0, %1, off"
                 :: "v"(lds_off), "v"(gsrc) : "memory");
}
static __device__ inline void wait_async() {
    asm volatile("s_wait_asynccnt 0x0" ::: "memory");
}

// ---------------------------------------------------------------------------
// Phase 1: brute-force KNN (K=3) with LDS tiling of the coarse point set.
// ---------------------------------------------------------------------------
#define KNN_TILE 1024

__global__ void knn_kernel(const float* __restrict__ pos,        // [N,3]
                           const int*   __restrict__ batch,      // [N]
                           const float* __restrict__ pos_skip,   // [Ns,3]
                           const int*   __restrict__ batch_skip, // [Ns]
                           int N, int Ns,
                           int*   __restrict__ out_idx,          // [Ns,3]
                           float* __restrict__ out_w) {          // [Ns,3] normalized
    __shared__ float sx[KNN_TILE], sy[KNN_TILE], sz[KNN_TILE];
    __shared__ int   sb[KNN_TILE];
    int t = blockIdx.x * blockDim.x + threadIdx.x;
    float qx = 0.f, qy = 0.f, qz = 0.f; int qb = -1;
    if (t < Ns) {
        qx = pos_skip[t * 3 + 0];
        qy = pos_skip[t * 3 + 1];
        qz = pos_skip[t * 3 + 2];
        qb = batch_skip[t];
    }
    float d0 = 1e30f, d1 = 1e30f, d2 = 1e30f;
    int   i0 = 0, i1 = 0, i2 = 0;
    for (int base = 0; base < N; base += KNN_TILE) {
        __syncthreads();
        for (int j = threadIdx.x; j < KNN_TILE; j += blockDim.x) {
            int g = base + j;
            if (g < N) {
                sx[j] = pos[g * 3 + 0];
                sy[j] = pos[g * 3 + 1];
                sz[j] = pos[g * 3 + 2];
                sb[j] = batch[g];
            } else {
                sx[j] = 0.f; sy[j] = 0.f; sz[j] = 0.f; sb[j] = -2;
            }
        }
        __syncthreads();
        int lim = min(KNN_TILE, N - base);
        for (int j = 0; j < lim; ++j) {
            float dx = sx[j] - qx, dy = sy[j] - qy, dz = sz[j] - qz;
            float d = dx * dx + dy * dy + dz * dz;
            d = (sb[j] == qb) ? d : 1e10f;   // BIG mask, matches reference
            int g = base + j;
            if (d < d2) {
                if (d < d0)      { d2 = d1; i2 = i1; d1 = d0; i1 = i0; d0 = d; i0 = g; }
                else if (d < d1) { d2 = d1; i2 = i1; d1 = d;  i1 = g; }
                else             { d2 = d;  i2 = g; }
            }
        }
    }
    if (t < Ns) {
        float w0 = 1.0f / fmaxf(d0, 1e-16f);
        float w1 = 1.0f / fmaxf(d1, 1e-16f);
        float w2 = 1.0f / fmaxf(d2, 1e-16f);
        float inv = 1.0f / (w0 + w1 + w2);
        out_idx[t * 3 + 0] = i0; out_idx[t * 3 + 1] = i1; out_idx[t * 3 + 2] = i2;
        out_w[t * 3 + 0] = w0 * inv; out_w[t * 3 + 1] = w1 * inv; out_w[t * 3 + 2] = w2 * inv;
    }
}

// ---------------------------------------------------------------------------
// Phase 2: weighted gather + skip-concat -> bf16 Y [Ns, 384], dword stores.
// Block = one skip row, 192 threads; each thread emits one packed bf16 pair.
// ---------------------------------------------------------------------------
__global__ void gather_kernel(const float* __restrict__ x,       // [N,256]
                              const float* __restrict__ x_skip,  // [Ns,128]
                              const int*   __restrict__ idx,     // [Ns,3]
                              const float* __restrict__ w,       // [Ns,3]
                              unsigned short* __restrict__ Ybf)  // [Ns,384] bf16
{
    int row = blockIdx.x;
    int i0 = idx[row * 3 + 0], i1 = idx[row * 3 + 1], i2 = idx[row * 3 + 2];
    float w0 = w[row * 3 + 0], w1 = w[row * 3 + 1], w2 = w[row * 3 + 2];
    unsigned int* Yrow = (unsigned int*)(Ybf + (size_t)row * 384);
    int c = threadIdx.x;  // 0..191
    if (c < 128) {
        int ch = c * 2;
        const float* p0 = x + (size_t)i0 * 256 + ch;
        const float* p1 = x + (size_t)i1 * 256 + ch;
        const float* p2 = x + (size_t)i2 * 256 + ch;
        float v0 = w0 * p0[0] + w1 * p1[0] + w2 * p2[0];
        float v1 = w0 * p0[1] + w1 * p1[1] + w2 * p2[1];
        Yrow[c] = pack_bf16x2(v0, v1);
    } else {
        int ch = (c - 128) * 2;
        const float* ps = x_skip + (size_t)row * 128 + ch;
        Yrow[c] = pack_bf16x2(ps[0], ps[1]);
    }
}

// ---------------------------------------------------------------------------
// Phase 3 prep: W [K,N] f32 -> Wt [N,K] bf16 (fragment-friendly: k contiguous).
// ---------------------------------------------------------------------------
__global__ void cvt_kernel(const float* __restrict__ W1, const float* __restrict__ W2,
                           unsigned short* __restrict__ W1t,   // [256,384]
                           unsigned short* __restrict__ W2t) { // [256,256]
    int i = blockIdx.x * blockDim.x + threadIdx.x;
    if (i < 384 * 256) {
        int n = i / 384, k = i - n * 384;
        W1t[i] = f32_to_bf16(W1[k * 256 + n]);
    }
    if (i < 256 * 256) {
        int n = i >> 8, k = i & 255;
        W2t[i] = f32_to_bf16(W2[k * 256 + n]);
    }
}

// ---------------------------------------------------------------------------
// Phase 3: fused GEMM + bias + ReLU. D = relu(A[M,K] * Wt[N,K]^T + b).
// Block: 256 threads / 8 waves, 128x64 block tile; each wave owns 32x32
// (2x2 WMMA tiles of v_wmma_f32_16x16x32_bf16). Tiles staged to LDS with
// global_load_async_to_lds_b128 + double buffering (ASYNCcnt).
// ---------------------------------------------------------------------------
template <int KDIM, bool OUT_BF16>
__global__ void gemm_relu_kernel(const unsigned short* __restrict__ A,   // [M,KDIM] bf16
                                 const unsigned short* __restrict__ Bt,  // [256,KDIM] bf16
                                 const float* __restrict__ bias,         // [256]
                                 void* __restrict__ Out) {
    constexpr int NDIM = 256;
    constexpr int STRIDE = 40;                 // ushorts/row: 80B -> 16B aligned rows
    constexpr int A_WORDS = 128 * STRIDE;      // A tile: 128 rows x 32 k
    constexpr int B_WORDS = 64 * STRIDE;       // B tile: 64 cols x 32 k
    __shared__ unsigned short lds[2][A_WORDS + B_WORDS];

    const int tid  = threadIdx.x;
    const int lane = tid & 31;
    const int wave = tid >> 5;
    const int wm = wave & 3;        // 0..3 -> 32-row slice
    const int wn = wave >> 2;       // 0..1 -> 32-col slice
    const int hi = (lane >> 4) & 1;
    const int r  = lane & 15;
    const int m0 = blockIdx.x * 128;
    const int n0 = blockIdx.y * 64;

    v8f acc[2][2];
#pragma unroll
    for (int mi = 0; mi < 2; ++mi)
#pragma unroll
        for (int ni = 0; ni < 2; ++ni)
            acc[mi][ni] = v8f{0.f, 0.f, 0.f, 0.f, 0.f, 0.f, 0.f, 0.f};

    // Async-stage one 32-wide k-slab (A: 8KB as 512 b128 chunks, B: 4KB as 256).
    auto stage = [&](unsigned short* buf, int kb) {
        unsigned short* Abuf = buf;
        unsigned short* Bbuf = buf + A_WORDS;
#pragma unroll
        for (int t = 0; t < 2; ++t) {
            int q = tid + t * 256;
            int row = q >> 2, seg = q & 3;
            async_copy_b128(&Abuf[row * STRIDE + seg * 8],
                            &A[(size_t)(m0 + row) * KDIM + kb + seg * 8]);
        }
        {
            int row = tid >> 2, seg = tid & 3;
            async_copy_b128(&Bbuf[row * STRIDE + seg * 8],
                            &Bt[(size_t)(n0 + row) * KDIM + kb + seg * 8]);
        }
    };

    stage(lds[0], 0);
    wait_async();
    __syncthreads();

    int cur = 0;
    for (int kb = 0; kb < KDIM; kb += 32) {
        if (kb + 32 < KDIM) stage(lds[cur ^ 1], kb + 32);   // prefetch next slab

        const unsigned short* Abuf = lds[cur];
        const unsigned short* Bbuf = lds[cur] + A_WORDS;

        // ISA 16-bit fragment layout: lane(0-15)=row/col r; VGPR v holds
        // K = {0..7,16..23} (+8 for the upper half-wave), pair-packed.
        Pack8 pa[2], pb[2];
#pragma unroll
        for (int mi = 0; mi < 2; ++mi) {
            int arow = wm * 32 + mi * 16 + r;
#pragma unroll
            for (int v = 0; v < 8; ++v) {
                int kbase = ((v < 4) ? 0 : 16) + (hi ? 8 : 0) + 2 * (v & 3);
                pa[mi].u[v] = *(const unsigned int*)&Abuf[arow * STRIDE + kbase];
            }
        }
#pragma unroll
        for (int ni = 0; ni < 2; ++ni) {
            int bcol = wn * 32 + ni * 16 + r;
#pragma unroll
            for (int v = 0; v < 8; ++v) {
                int kbase = ((v < 4) ? 0 : 16) + (hi ? 8 : 0) + 2 * (v & 3);
                pb[ni].u[v] = *(const unsigned int*)&Bbuf[bcol * STRIDE + kbase];
            }
        }
#pragma unroll
        for (int mi = 0; mi < 2; ++mi) {
            v16bf af = __builtin_bit_cast(v16bf, pa[mi]);
#pragma unroll
            for (int ni = 0; ni < 2; ++ni) {
                v16bf bfrag = __builtin_bit_cast(v16bf, pb[ni]);
                acc[mi][ni] = __builtin_amdgcn_wmma_f32_16x16x32_bf16(
                    false, af, false, bfrag, (short)0, acc[mi][ni], false, false);
            }
        }
        wait_async();       // next slab landed (and our LDS reads precede barrier)
        __syncthreads();
        cur ^= 1;
    }

    // Epilogue: C/D layout — VGPR i: row = +8*hi + i, col = +r.
#pragma unroll
    for (int mi = 0; mi < 2; ++mi)
#pragma unroll
        for (int ni = 0; ni < 2; ++ni)
#pragma unroll
            for (int i = 0; i < 8; ++i) {
                int row = m0 + wm * 32 + mi * 16 + hi * 8 + i;
                int col = n0 + wn * 32 + ni * 16 + r;
                float val = fmaxf(acc[mi][ni][i] + bias[col], 0.0f);
                if (OUT_BF16) ((unsigned short*)Out)[(size_t)row * NDIM + col] = f32_to_bf16(val);
                else          ((float*)Out)[(size_t)row * NDIM + col] = val;
            }
}

// ---------------------------------------------------------------------------
// Tail: copy pos_skip (float) and batch_skip (int bits) into concatenated out.
// ---------------------------------------------------------------------------
__global__ void tail_kernel(const float* __restrict__ pos_skip,
                            const int* __restrict__ batch_skip,
                            float* __restrict__ outPos, int* __restrict__ outBatch, int Ns) {
    int i = blockIdx.x * blockDim.x + threadIdx.x;
    if (i < Ns * 3) outPos[i] = pos_skip[i];
    if (i < Ns)     outBatch[i] = batch_skip[i];
}

extern "C" void kernel_launch(void* const* d_in, const int* in_sizes, int n_in,
                              void* d_out, int out_size, void* d_ws, size_t ws_size,
                              hipStream_t stream) {
    const float* x          = (const float*)d_in[0];
    const float* pos        = (const float*)d_in[1];
    const int*   batch      = (const int*)d_in[2];
    const float* x_skip     = (const float*)d_in[3];
    const float* pos_skip   = (const float*)d_in[4];
    const int*   batch_skip = (const int*)d_in[5];
    const float* W1         = (const float*)d_in[6];
    const float* b1         = (const float*)d_in[7];
    const float* W2         = (const float*)d_in[8];
    const float* b2         = (const float*)d_in[9];
    (void)n_in; (void)out_size; (void)ws_size;

    const int N  = in_sizes[1] / 3;   // 4096
    const int Ns = in_sizes[4] / 3;   // 16384

    // Workspace carve (256B-aligned regions).
    char* ws = (char*)d_ws;
    size_t off = 0;
    auto carve = [&](size_t bytes) -> void* {
        void* p = ws + off;
        off = (off + bytes + 255) & ~(size_t)255;
        return p;
    };
    int*            idxBuf = (int*)           carve((size_t)Ns * 3 * sizeof(int));
    float*          wBuf   = (float*)         carve((size_t)Ns * 3 * sizeof(float));
    unsigned short* Ybf    = (unsigned short*)carve((size_t)Ns * 384 * 2);
    unsigned short* W1t    = (unsigned short*)carve((size_t)256 * 384 * 2);
    unsigned short* W2t    = (unsigned short*)carve((size_t)256 * 256 * 2);
    unsigned short* Hbf    = (unsigned short*)carve((size_t)Ns * 256 * 2);

    knn_kernel<<<(Ns + 255) / 256, 256, 0, stream>>>(pos, batch, pos_skip, batch_skip,
                                                     N, Ns, idxBuf, wBuf);
    gather_kernel<<<Ns, 192, 0, stream>>>(x, x_skip, idxBuf, wBuf, Ybf);
    cvt_kernel<<<(384 * 256 + 255) / 256, 256, 0, stream>>>(W1, W2, W1t, W2t);

    gemm_relu_kernel<384, true><<<dim3(Ns / 128, 256 / 64), 256, 0, stream>>>(Ybf, W1t, b1, Hbf);

    float* outMain = (float*)d_out;
    gemm_relu_kernel<256, false><<<dim3(Ns / 128, 256 / 64), 256, 0, stream>>>(Hbf, W2t, b2, outMain);

    float* outPos   = outMain + (size_t)Ns * 256;
    int*   outBatch = (int*)(outPos + (size_t)Ns * 3);
    tail_kernel<<<(Ns * 3 + 255) / 256, 256, 0, stream>>>(pos_skip, batch_skip, outPos, outBatch, Ns);
}
